// MultiHeadCrossAttention_53618371723270
// MI455X (gfx1250) — compile-verified, matrix operand order checked by reading, not run on hardware
//
#include <hip/hip_runtime.h>
#include <cstdint>
#include <cstddef>

// ---------------------------------------------------------------------------
// CDNA5 (gfx1250) multi-head cross-attention, bf16 WMMA everywhere.
//   B=4, Nq=1024, Nk=2048, model=1024, keydim=768, H=16, D=64
// ---------------------------------------------------------------------------

typedef __attribute__((ext_vector_type(16))) __bf16 v16bf;
typedef __attribute__((ext_vector_type(8)))  float  v8f;

union FragU {
    v16bf v;
    struct { uint4 lo, hi; } q;
};

union PackU {
    __bf16 h[8];
    uint4  u;
};

// Two 16-byte chunks -> one 16-element bf16 fragment (CDNA5 per-lane layout).
__device__ __forceinline__ v16bf load_frag(const __bf16* p0, const __bf16* p1) {
    FragU u;
    u.q.lo = *reinterpret_cast<const uint4*>(p0);
    u.q.hi = *reinterpret_cast<const uint4*>(p1);
    return u.v;
}

__device__ __forceinline__ v8f wmma_bf16(v16bf a, v16bf b, v8f c) {
    return __builtin_amdgcn_wmma_f32_16x16x32_bf16(false, a, false, b,
                                                   (short)0, c, false, false);
}

// Exchange a value with the partner lane in the other 16-lane half (xor 16).
__device__ __forceinline__ float xhalf(float x) {
#if __has_builtin(__builtin_amdgcn_permlanex16)
    int i = __float_as_int(x);
    int r = __builtin_amdgcn_permlanex16(i, i, 0x76543210, 0xfedcba98,
                                         false, false);
    return __int_as_float(r);
#else
    return __shfl_xor(x, 16, 32);
#endif
}

// ---------------------------------------------------------------------------
// Tiled GEMM:  C[M,N] = A[M,K] @ W[N,K]^T + bias[N]
//   block tile 128x128, K-step 64, 256 threads = 8 wave32s, wave tile 32x64.
// ---------------------------------------------------------------------------
template <typename TA, typename TB, bool OUT_F32, bool TRANS>
__global__ __launch_bounds__(256)
void gemm_wmma(const TA* __restrict__ A, const TB* __restrict__ W,
               const float* __restrict__ bias, void* __restrict__ outp,
               int M, int N, int K, int rows_per_batch)
{
    constexpr int BM = 128, BN = 128, BK = 64, PADS = 72; // 72 bf16 = 144B rows
    __shared__ __bf16 sA[BM * PADS];
    __shared__ __bf16 sB[BN * PADS];

    const int tid  = threadIdx.x;
    const int lane = tid & 31;
    const int wid  = tid >> 5;
    const int wm   = wid & 3;          // 4 waves along M  (4 * 32 = 128)
    const int wn   = wid >> 2;         // 2 waves along N  (2 * 64 = 128)
    const int m0   = blockIdx.y * BM;
    const int n0   = blockIdx.x * BN;

    v8f acc[2][4];
#pragma unroll
    for (int i = 0; i < 2; ++i)
#pragma unroll
        for (int j = 0; j < 4; ++j) acc[i][j] = {};

    // staging role: each thread fills one 32-element half-row of each tile
    const int srow  = tid >> 1;
    const int shalf = tid & 1;
    const TA* aSrc = A + (size_t)(m0 + srow) * K + shalf * 32;
    const TB* bSrc = W + (size_t)(n0 + srow) * K + shalf * 32;
    __bf16*   aDst = sA + srow * PADS + shalf * 32;
    __bf16*   bDst = sB + srow * PADS + shalf * 32;

    const int fr = lane & 15;          // fragment row/col within 16
    const int ks = (lane >> 4) & 1;    // K-half selector

    for (int k0 = 0; k0 < K; k0 += BK) {
#pragma unroll
        for (int i = 0; i < 32; ++i) aDst[i] = (__bf16)aSrc[i];
#pragma unroll
        for (int i = 0; i < 32; ++i) bDst[i] = (__bf16)bSrc[i];
        if (k0 + BK < K) {             // -> global_prefetch_b8
            __builtin_prefetch(aSrc + BK, 0, 1);
            __builtin_prefetch(bSrc + BK, 0, 1);
        }
        aSrc += BK;
        bSrc += BK;
        __syncthreads();

#pragma unroll
        for (int kc = 0; kc < 2; ++kc) {
            v16bf af[2], bf[4];
#pragma unroll
            for (int i = 0; i < 2; ++i) {
                const __bf16* p = sA + (wm * 32 + i * 16 + fr) * PADS +
                                  kc * 32 + ks * 8;
                af[i] = load_frag(p, p + 16);
            }
#pragma unroll
            for (int j = 0; j < 4; ++j) {
                const __bf16* p = sB + (wn * 64 + j * 16 + fr) * PADS +
                                  kc * 32 + ks * 8;
                bf[j] = load_frag(p, p + 16);
            }
#pragma unroll
            for (int i = 0; i < 2; ++i)
#pragma unroll
                for (int j = 0; j < 4; ++j)
                    acc[i][j] = wmma_bf16(af[i], bf[j], acc[i][j]);
        }
        __syncthreads();
    }

    // store: C element (reg r, lane) -> row = r + 8*(lane>>4), col = lane&15
#pragma unroll
    for (int i = 0; i < 2; ++i)
#pragma unroll
        for (int j = 0; j < 4; ++j)
#pragma unroll
            for (int r = 0; r < 8; ++r) {
                const int mrow = m0 + wm * 32 + i * 16 + r + 8 * ks;
                const int ncol = n0 + wn * 64 + j * 16 + fr;
                const float v  = acc[i][j][r] + bias[ncol];
                if constexpr (TRANS) {
                    const int bb = mrow / rows_per_batch;
                    const int tt = mrow % rows_per_batch;
                    const size_t idx = (size_t)bb * N * rows_per_batch +
                                       (size_t)ncol * rows_per_batch + tt;
                    ((__bf16*)outp)[idx] = (__bf16)v;
                } else if constexpr (OUT_F32) {
                    ((float*)outp)[(size_t)mrow * N + ncol] = v;
                } else {
                    ((__bf16*)outp)[(size_t)mrow * N + ncol] = (__bf16)v;
                }
            }
}

// ---------------------------------------------------------------------------
// Flash attention, fully transposed, zero LDS, software-pipelined.
// One block per (q-tile 128, head, batch); 8 wave32s, wave owns 16 queries.
//   S^T(32k x 16q) = K(32x64) Q^T          (keys on registers, queries on lanes)
//   -> in-lane softmax (one permlanex16 half swap per reduction)
//   -> S^T C-layout registers ARE the P^T B-fragment (no data movement)
//   O^T(64d x 16q) += V^T(64x32) P^T(32x16)
// K fragments are double-buffered (2x-unrolled ping-pong) so the next block's
// global loads fly while the softmax VALU/trans work executes.
// ---------------------------------------------------------------------------
__global__ __launch_bounds__(256)
void flash_attn(const __bf16* __restrict__ Qb, const __bf16* __restrict__ Kb,
                const __bf16* __restrict__ Vt, __bf16* __restrict__ Ob,
                int B, int H, int Nq, int Nk, int Dm)
{
    constexpr int D = 64;
    const int tid  = threadIdx.x;
    const int lane = tid & 31;
    const int wid  = tid >> 5;
    const int h    = blockIdx.y;
    const int b    = blockIdx.z;
    const int q0   = blockIdx.x * 128 + wid * 16;
    const int fr   = lane & 15;        // query (lanes), or key/d row (A frags)
    const int ks   = (lane >> 4) & 1;
    const float scale = 0.125f;        // 1/sqrt(64)

    // Q^T B-fragments: lane = query, two K-steps over D=64
    const __bf16* qr = Qb + (size_t)(b * Nq + q0 + fr) * Dm + h * D + ks * 8;
    const v16bf qf0 = load_frag(qr,      qr + 16);
    const v16bf qf1 = load_frag(qr + 32, qr + 48);

    // running pointers (constant strides; block offsets fold into inst offsets)
    const __bf16* kptr = Kb + (size_t)(b * Nk + fr) * Dm + h * D + ks * 8;
    const __bf16* vptr = Vt + ((size_t)b * Dm + h * D + fr) * Nk + ks * 8;
    const int kStep = 32 * Dm;         // 32 keys per iteration

    v8f ot[4];
#pragma unroll
    for (int j = 0; j < 4; ++j) ot[j] = {};
    float m = -3.0e38f, l = 0.0f;

    auto loadK4 = [&](const __bf16* kp, v16bf kf[4]) {
        const __bf16* k1 = kp + 16 * Dm;
        kf[0] = load_frag(kp,      kp + 16);
        kf[1] = load_frag(kp + 32, kp + 48);
        kf[2] = load_frag(k1,      k1 + 16);
        kf[3] = load_frag(k1 + 32, k1 + 48);
    };

    // one 32-key step: consumes kcur, prefetches the block at knext_ptr
    auto step = [&](const v16bf kcur[4], v16bf knext[4],
                    const __bf16* knext_ptr, bool preload) {
        // V fragments for this block — issue early, consumed after softmax
        v16bf vf[4];
#pragma unroll
        for (int j = 0; j < 4; ++j) {
            const __bf16* vr = vptr + j * 16 * Nk;
            vf[j] = load_frag(vr, vr + 16);
        }

        // ----- S^T = K Q^T : two 16-key M blocks -----
        v8f st0 = {}, st1 = {};
        st0 = wmma_bf16(kcur[0], qf0, st0);
        st0 = wmma_bf16(kcur[1], qf1, st0);
        st1 = wmma_bf16(kcur[2], qf0, st1);
        st1 = wmma_bf16(kcur[3], qf1, st1);

        // next block's K loads fly while the softmax below executes
        if (preload) loadK4(knext_ptr, knext);

        // ----- online softmax: all state per-lane (per-query) -----
        float mx = -3.0e38f;
#pragma unroll
        for (int r = 0; r < 8; ++r) {
            st0[r] *= scale; st1[r] *= scale;
            mx = fmaxf(mx, fmaxf(st0[r], st1[r]));
        }
        mx = fmaxf(mx, xhalf(mx));                 // other 16 keys
        const float mnew = fmaxf(m, mx);
        const float corr = __expf(m - mnew);
        float rs = 0.0f;
#pragma unroll
        for (int r = 0; r < 8; ++r) {
            st0[r] = __expf(st0[r] - mnew); rs += st0[r];
            st1[r] = __expf(st1[r] - mnew); rs += st1[r];
        }
        rs += xhalf(rs);
        l = l * corr + rs;
        m = mnew;
#pragma unroll
        for (int j = 0; j < 4; ++j)
#pragma unroll
            for (int r = 0; r < 8; ++r) ot[j][r] *= corr;

        // ----- S^T registers are already the P^T B-fragment -----
        v16bf pf;
#pragma unroll
        for (int r = 0; r < 8; ++r) {
            pf[r]     = (__bf16)st0[r];
            pf[r + 8] = (__bf16)st1[r];
        }

        // ----- O^T += V^T P^T : four 16-d M blocks -----
#pragma unroll
        for (int j = 0; j < 4; ++j)
            ot[j] = wmma_bf16(vf[j], pf, ot[j]);

        vptr += 32;
    };

    // 2x-unrolled ping-pong over K fragment buffers (Nk/32 is even)
    v16bf kA[4], kB[4];
    loadK4(kptr, kA);
    for (int k0 = 0; k0 < Nk; k0 += 64) {
        step(kA, kB, kptr + kStep, true);                    // k0 .. k0+31
        step(kB, kA, kptr + 2 * kStep, k0 + 64 < Nk);        // k0+32 .. k0+63
        kptr += 2 * kStep;
    }

    // ----- finalize: per-lane 1/l, 8 contiguous d per register block -----
    const float inv = 1.0f / l;
    __bf16* orow = Ob + (size_t)(b * Nq + q0 + fr) * Dm + h * D + 8 * ks;
#pragma unroll
    for (int j = 0; j < 4; ++j) {
        PackU p;
#pragma unroll
        for (int r = 0; r < 8; ++r) p.h[r] = (__bf16)(ot[j][r] * inv);
        *reinterpret_cast<uint4*>(orow + j * 16) = p.u;
    }
}

// ---------------------------------------------------------------------------
extern "C" void kernel_launch(void* const* d_in, const int* in_sizes, int n_in,
                              void* d_out, int out_size, void* d_ws, size_t ws_size,
                              hipStream_t stream)
{
    (void)in_sizes; (void)n_in; (void)out_size; (void)ws_size;
    const float* query = (const float*)d_in[0];
    const float* key   = (const float*)d_in[1];
    const float* value = (const float*)d_in[2];
    const float* Wq    = (const float*)d_in[3];
    const float* bq    = (const float*)d_in[4];
    const float* Wk    = (const float*)d_in[5];
    const float* bk    = (const float*)d_in[6];
    const float* Wv    = (const float*)d_in[7];
    const float* bv    = (const float*)d_in[8];
    const float* Wo    = (const float*)d_in[9];
    const float* bo    = (const float*)d_in[10];

    constexpr int B = 4, Nq = 1024, Nk = 2048, Dm = 1024, Kd = 768, H = 16;

    char* ws = (char*)d_ws;
    const size_t szQ = (size_t)B * Nq * Dm * sizeof(__bf16);   //  8 MiB
    const size_t szK = (size_t)B * Nk * Dm * sizeof(__bf16);   // 16 MiB
    __bf16* Qb = (__bf16*)(ws);
    __bf16* Kb = (__bf16*)(ws + szQ);
    __bf16* Vt = (__bf16*)(ws + szQ + szK);
    __bf16* Ob = (__bf16*)(ws + szQ + 2 * szK);                // total 48 MiB

    const dim3 blk(256);

    // Q/K/V projections (fp32 -> bf16 WMMA -> bf16 out; V stored transposed)
    gemm_wmma<float, float, false, false>
        <<<dim3(Dm / 128, (B * Nq) / 128), blk, 0, stream>>>(
            query, Wq, bq, Qb, B * Nq, Dm, Dm, 1);
    gemm_wmma<float, float, false, false>
        <<<dim3(Dm / 128, (B * Nk) / 128), blk, 0, stream>>>(
            key, Wk, bk, Kb, B * Nk, Dm, Kd, 1);
    gemm_wmma<float, float, false, true>
        <<<dim3(Dm / 128, (B * Nk) / 128), blk, 0, stream>>>(
            value, Wv, bv, Vt, B * Nk, Dm, Kd, Nk);

    // Streaming softmax attention (register-resident, no LDS, pipelined)
    flash_attn<<<dim3(Nq / 128, H, B), blk, 0, stream>>>(
        Qb, Kb, Vt, Ob, B, H, Nq, Nk, Dm);

    // Output projection (bf16 -> fp32 + bias)
    gemm_wmma<__bf16, float, true, false>
        <<<dim3(Dm / 128, (B * Nq) / 128), blk, 0, stream>>>(
            Ob, Wo, bo, d_out, B * Nq, Dm, Dm, 1);
}
